// MoEGPTBlock_56298431316471
// MI455X (gfx1250) — matrix operand order repaired and checked by reading, not compiled, not run on hardware
//
#include <hip/hip_runtime.h>

// ---------------- problem constants ----------------
constexpr int kB  = 2;
constexpr int kS  = 2048;
constexpr int kH  = 1024;
constexpr int kNH = 16;
constexpr int kDH = 64;
constexpr int kE  = 8;
constexpr int kF  = 4096;
constexpr int kT  = kB * kS;          // 4096 tokens
constexpr int k3H = 3 * kH;
constexpr int kFC = 128;              // MoE F-chunk width

typedef __attribute__((ext_vector_type(16))) _Float16 v16h;
typedef __attribute__((ext_vector_type(8)))  float    v8f;

// ---------------- WMMA helper (CDNA5 16x16x32 f16 -> f32) ----------------
__device__ __forceinline__ v8f wmma_f16(v16h a, v16h b, v8f c) {
  return __builtin_amdgcn_wmma_f32_16x16x32_f16(false, a, false, b, (short)0, c,
                                                false, false);
}

// A fragment: 16x32, row-major f16 source, base already at (m0,k0). Contiguous k order (not ISA table order):
// lane m = lane&15; element e -> K = 16*(lane>>4) + e
__device__ __forceinline__ v16h load_a_rm(const _Float16* A, int lda, int lane) {
  v16h a;
  int m  = lane & 15;
  int kb = (lane >> 4) << 4;
  const _Float16* p = A + (size_t)m * lda;
#pragma unroll
  for (int e = 0; e < 16; ++e) {
    int k = kb + e;
    a[e] = p[k];
  }
  return a;
}

// B fragment where memory holds B^T row-major (W[N,Kd] f16, B(k,n)=W[n][k]).
// lane n = lane&15; K = 16*(lane>>4)+e -> one contiguous 32B run per lane.
__device__ __forceinline__ v16h load_b_nt_f16(const _Float16* W, int ldw, int lane) {
  v16h b;
  int n  = lane & 15;
  int kb = (lane >> 4) << 4;
  const _Float16* p = W + (size_t)n * ldw + kb;
#pragma unroll
  for (int e = 0; e < 16; ++e) b[e] = p[e];
  return b;
}

// B fragment where memory holds B row-major (W[Kd,N] f16, B(k,n)=W[k][n]).
__device__ __forceinline__ v16h load_b_rm_f16(const _Float16* W, int ldw, int lane) {
  v16h b;
  int n  = lane & 15;
  int kb = (lane >> 4) << 4;
#pragma unroll
  for (int e = 0; e < 16; ++e) b[e] = W[(size_t)(kb + e) * ldw + n];
  return b;
}

// ---------------- weight conversion kernels (run once per launch) ----------
__global__ __launch_bounds__(256) void cvt_kernel(const float* __restrict__ in,
                                                  _Float16* __restrict__ out,
                                                  int n) {
  int i = (blockIdx.x * 256 + threadIdx.x) * 4;
  if (i + 3 < n) {
#pragma unroll
    for (int j = 0; j < 4; ++j) out[i + j] = (_Float16)in[i + j];
  } else {
    for (int j = 0; j < 4 && i + j < n; ++j) out[i + j] = (_Float16)in[i + j];
  }
}

// transpose+convert: in[e][R][C] f32 -> out[e][C][R] f16  (32x32 LDS tiles)
__global__ __launch_bounds__(256) void transpose_cvt_kernel(
    const float* __restrict__ in, _Float16* __restrict__ outT, int R, int C) {
  __shared__ _Float16 tile[32][33];
  int e = blockIdx.z;
  const float* I = in + (size_t)e * R * C;
  _Float16*    O = outT + (size_t)e * R * C;
  int c0 = blockIdx.x * 32, r0 = blockIdx.y * 32;
  int tx = threadIdx.x & 31, ty = threadIdx.x >> 5;   // 8 rows of 32
#pragma unroll
  for (int i = 0; i < 32; i += 8)
    tile[ty + i][tx] = (_Float16)I[(size_t)(r0 + ty + i) * C + c0 + tx];
  __syncthreads();
#pragma unroll
  for (int i = 0; i < 32; i += 8)
    O[(size_t)(c0 + ty + i) * R + r0 + tx] = tile[tx][ty + i];
}

// ---------------- LayerNorm -> f16 ----------------
__global__ __launch_bounds__(256) void ln_kernel(const float* __restrict__ x,
                                                 const float* __restrict__ g,
                                                 const float* __restrict__ b,
                                                 _Float16* __restrict__ out) {
  __shared__ float red[256];
  int t = blockIdx.x, tid = threadIdx.x;
  const float* xr = x + (size_t)t * kH;
  float lv[4], s = 0.f;
#pragma unroll
  for (int i = 0; i < 4; ++i) { lv[i] = xr[tid + i * 256]; s += lv[i]; }
  red[tid] = s; __syncthreads();
  for (int off = 128; off; off >>= 1) {
    if (tid < off) red[tid] += red[tid + off];
    __syncthreads();
  }
  float mu = red[0] * (1.f / kH); __syncthreads();
  float s2 = 0.f;
#pragma unroll
  for (int i = 0; i < 4; ++i) { float d = lv[i] - mu; s2 += d * d; }
  red[tid] = s2; __syncthreads();
  for (int off = 128; off; off >>= 1) {
    if (tid < off) red[tid] += red[tid + off];
    __syncthreads();
  }
  float rstd = rsqrtf(red[0] * (1.f / kH) + 1e-5f);
#pragma unroll
  for (int i = 0; i < 4; ++i) {
    int c = tid + i * 256;
    out[(size_t)t * kH + c] = (_Float16)((lv[i] - mu) * rstd * g[c] + b[c]);
  }
}

// ---------------- generic GEMM: C = A(f16,[M,Kd]) * W^T(f16,[N,Kd]) + bias ---
// optional residual add, f32 and/or f16 outputs. block = 64M x 32N (8 waves).
__global__ __launch_bounds__(256) void gemm_kernel(const _Float16* __restrict__ A,
                                                   const _Float16* __restrict__ W,
                                                   const float* __restrict__ bias,
                                                   const float* __restrict__ residual,
                                                   float* __restrict__ outf,
                                                   _Float16* __restrict__ outh,
                                                   int M, int N, int Kd) {
  int lane = threadIdx.x & 31, wave = threadIdx.x >> 5;
  int m0 = blockIdx.y * 64 + (wave & 3) * 16;
  int n0 = blockIdx.x * 32 + (wave >> 2) * 16;
  const _Float16* Ap = A + (size_t)m0 * Kd;
  const _Float16* Wp = W + (size_t)n0 * Kd;
  v8f acc = {};
  for (int k0 = 0; k0 < Kd; k0 += 32) {
    if (k0 + 64 < Kd) {   // prefetch two K-tiles ahead -> global_prefetch_b8
      __builtin_prefetch(Ap + (lane & 15) * Kd + k0 + 64, 0, 1);
      __builtin_prefetch(Wp + (lane & 15) * Kd + k0 + 64, 0, 1);
    }
    v16h a = load_a_rm(Ap + k0, Kd, lane);
    v16h b = load_b_nt_f16(Wp + k0, Kd, lane);
    acc = wmma_f16(a, b, acc);
  }
  int n = lane & 15, mb = (lane >> 4) << 3;
  float bn = bias[n0 + n];
#pragma unroll
  for (int r = 0; r < 8; ++r) {
    size_t idx = (size_t)(m0 + mb + r) * N + (n0 + n);
    float v = acc[r] + bn;
    if (residual) v += residual[idx];
    if (outf) outf[idx] = v;
    if (outh) outh[idx] = (_Float16)v;
  }
}

// ---------------- attention: per (qtile16, head, batch) ----------------
__global__ __launch_bounds__(256) void attn_kernel(const _Float16* __restrict__ qkv,
                                                   _Float16* __restrict__ o) {
  extern __shared__ char smem[];
  float*    sc   = (float*)smem;                          // [16][kS] f32
  _Float16* pbuf = (_Float16*)(smem + 16 * kS * 4);       // [16][kS] f16
  __shared__ float r16s[16][16];
  __shared__ float rmax[16], rinv[16];

  int qt = blockIdx.x, hh = blockIdx.y, bb = blockIdx.z;
  int lane = threadIdx.x & 31, wave = threadIdx.x >> 5;
  size_t base = (size_t)bb * kS * k3H;
  const _Float16* q  = qkv + base + (size_t)(qt * 16) * k3H + hh * kDH;
  const _Float16* kk = qkv + base + kH     + hh * kDH;
  const _Float16* vv = qkv + base + 2 * kH + hh * kDH;

  // phase 1: scores = q . k^T / sqrt(DH)
  for (int kt = wave; kt < kS / 16; kt += 8) {
    v8f acc = {};
    for (int k0 = 0; k0 < kDH; k0 += 32) {
      v16h a = load_a_rm(q + k0, k3H, lane);
      v16h b = load_b_nt_f16(kk + (size_t)(kt * 16) * k3H + k0, k3H, lane);
      acc = wmma_f16(a, b, acc);
    }
    int n = lane & 15, mb = (lane >> 4) << 3;
#pragma unroll
    for (int r = 0; r < 8; ++r)
      sc[(mb + r) * kS + kt * 16 + n] = acc[r] * 0.125f;
  }
  __syncthreads();

  // phase 2: row softmax (unnormalized exp to f16, 1/sum saved)
  int row = threadIdx.x >> 4, sub = threadIdx.x & 15;
  float mx = -1e30f;
  for (int j = sub; j < kS; j += 16) mx = fmaxf(mx, sc[row * kS + j]);
  r16s[row][sub] = mx; __syncthreads();
  if (sub == 0) {
    float m2 = -1e30f;
    for (int i = 0; i < 16; ++i) m2 = fmaxf(m2, r16s[row][i]);
    rmax[row] = m2;
  }
  __syncthreads();
  float mrow = rmax[row], sum = 0.f;
  for (int j = sub; j < kS; j += 16) {
    float e = __expf(sc[row * kS + j] - mrow);
    pbuf[row * kS + j] = (_Float16)e;
    sum += e;
  }
  r16s[row][sub] = sum; __syncthreads();
  if (sub == 0) {
    float s2 = 0.f;
    for (int i = 0; i < 16; ++i) s2 += r16s[row][i];
    rinv[row] = 1.f / s2;
  }
  __syncthreads();

  // phase 3: o = (P * V) * rinv   (4 waves, one 16-col tile of DH each)
  if (wave < 4) {
    int nt = wave;
    v8f acc = {};
    for (int kt = 0; kt < kS / 32; ++kt) {
      v16h a = load_a_rm(pbuf + kt * 32, kS, lane);
      v16h b = load_b_rm_f16(vv + (size_t)(kt * 32) * k3H + nt * 16, k3H, lane);
      acc = wmma_f16(a, b, acc);
    }
    int n = lane & 15, mb = (lane >> 4) << 3;
#pragma unroll
    for (int r = 0; r < 8; ++r) {
      int m = mb + r;
      int tok = qt * 16 + m;
      o[((size_t)(bb * kS + tok)) * kH + hh * kDH + nt * 16 + n] =
          (_Float16)(acc[r] * rinv[m]);
    }
  }
}

// ---------------- router: logits, top-2, renorm softmax, append -------------
__global__ __launch_bounds__(256) void router_kernel(const _Float16* __restrict__ h2,
                                                     const float* __restrict__ Wg,
                                                     int* __restrict__ cnt,
                                                     int* __restrict__ list,
                                                     float* __restrict__ gw) {
  __shared__ float red[256 * 8];
  int t = blockIdx.x, tid = threadIdx.x;
  float acc[8] = {};
  for (int h = tid; h < kH; h += 256) {
    float xv = (float)h2[(size_t)t * kH + h];
    const float* wr = Wg + (size_t)h * kE;
#pragma unroll
    for (int e = 0; e < 8; ++e) acc[e] += xv * wr[e];
  }
#pragma unroll
  for (int e = 0; e < 8; ++e) red[tid * 8 + e] = acc[e];
  __syncthreads();
  for (int off = 128; off; off >>= 1) {
    if (tid < off)
#pragma unroll
      for (int e = 0; e < 8; ++e) red[tid * 8 + e] += red[(tid + off) * 8 + e];
    __syncthreads();
  }
  if (tid == 0) {
    float lg[8];
#pragma unroll
    for (int e = 0; e < 8; ++e) lg[e] = red[e];
    int i0 = 0;
    for (int e = 1; e < 8; ++e) if (lg[e] > lg[i0]) i0 = e;
    int i1 = -1;
    for (int e = 0; e < 8; ++e)
      if (e != i0 && (i1 < 0 || lg[e] > lg[i1])) i1 = e;
    float e1 = __expf(lg[i1] - lg[i0]);
    float w0 = 1.f / (1.f + e1), w1 = e1 / (1.f + e1);
    int p0 = atomicAdd(&cnt[i0], 1);
    list[i0 * kT + p0] = t * 2 + 0; gw[i0 * kT + p0] = w0;
    int p1 = atomicAdd(&cnt[i1], 1);
    list[i1 * kT + p1] = t * 2 + 1; gw[i1 * kT + p1] = w1;
  }
}

__global__ void zero_cnt_kernel(int* cnt) {
  if (threadIdx.x < kE) cnt[threadIdx.x] = 0;
}

// ---------------- fused MoE expert kernel ----------------
// grid = (T/16 tiles, E). F processed in 128-wide chunks; both GEMM phases use
// all 8 waves (32 WMMA/wave per phase per chunk).
// W1T: [E][F][H] f16 (B^T form for GEMM1), W2T: [E][H][F] f16 (B^T for GEMM2).
__global__ __launch_bounds__(256) void moe_kernel(const _Float16* __restrict__ h2,
                                                  const _Float16* __restrict__ W1T,
                                                  const float* __restrict__ b1,
                                                  const _Float16* __restrict__ W2T,
                                                  const float* __restrict__ b2,
                                                  const int* __restrict__ cnt,
                                                  const int* __restrict__ list,
                                                  const float* __restrict__ gw,
                                                  float* __restrict__ contrib) {
  extern __shared__ char smem[];
  _Float16* Af   = (_Float16*)smem;                                  // 16*kH  f16
  float*    yacc = (float*)(smem + 16 * kH * 2);                     // 16*kH  f32
  _Float16* hid  = (_Float16*)(smem + 16 * kH * 2 + 16 * kH * 4);    // 16*kFC f16
  __shared__ int   toks[16];
  __shared__ float gts[16];

  int e = blockIdx.y;
  int nTok = cnt[e];
  int t0 = blockIdx.x * 16;
  if (t0 >= nTok) return;
  int tid = threadIdx.x, lane = tid & 31, wave = tid >> 5;

  if (tid < 16) {
    int idx = t0 + tid;
    if (idx < nTok) { toks[tid] = list[e * kT + idx]; gts[tid] = gw[e * kT + idx]; }
    else            { toks[tid] = -1;                 gts[tid] = 0.f; }
  }
  __syncthreads();

  for (int i = tid; i < 16 * kH; i += 256) {
    int r = i >> 10, c = i & (kH - 1);
    int entry = toks[r];
    Af[i]   = (entry >= 0) ? h2[(size_t)(entry >> 1) * kH + c] : (_Float16)0.f;
    yacc[i] = 0.f;
  }
  __syncthreads();

  const _Float16* W1e = W1T + (size_t)e * kH * kF;   // [F][H]
  const _Float16* W2e = W2T + (size_t)e * kF * kH;   // [H][F]

  for (int fc = 0; fc < kF; fc += kFC) {
    // phase 1: hid[16,kFC] = gelu(A * W1chunk + b1); wave w owns N-tile w
    {
      int nt = wave;
      v8f acc = {};
      const _Float16* Wp = W1e + (size_t)(fc + nt * 16) * kH;
      for (int k0 = 0; k0 < kH; k0 += 32) {
        if (k0 + 64 < kH)
          __builtin_prefetch(Wp + (size_t)(lane & 15) * kH + k0 + 64, 0, 1);
        v16h a = load_a_rm(Af + k0, kH, lane);
        v16h b = load_b_nt_f16(Wp + k0, kH, lane);
        acc = wmma_f16(a, b, acc);
      }
      int n = lane & 15, mb = (lane >> 4) << 3;
#pragma unroll
      for (int r = 0; r < 8; ++r) {
        float v  = acc[r] + b1[e * kF + fc + nt * 16 + n];
        float u  = 0.7978845608f * (v + 0.044715f * v * v * v);
        float ge = 0.5f * v * (1.f + tanhf(u));
        hid[(mb + r) * kFC + nt * 16 + n] = (_Float16)ge;
      }
    }
    __syncthreads();
    // phase 2: y[16,H] += hid * W2chunk; 64 N-tiles over 8 waves, Kd=kFC
    for (int nt2 = wave; nt2 < kH / 16; nt2 += 8) {
      int n = lane & 15, mb = (lane >> 4) << 3;
      v8f acc;
#pragma unroll
      for (int r = 0; r < 8; ++r) acc[r] = yacc[(mb + r) * kH + nt2 * 16 + n];
      const _Float16* Wp = W2e + (size_t)(nt2 * 16) * kF + fc;
#pragma unroll
      for (int k0 = 0; k0 < kFC; k0 += 32) {
        v16h a = load_a_rm(hid + k0, kFC, lane);
        v16h b = load_b_nt_f16(Wp + k0, kF, lane);
        acc = wmma_f16(a, b, acc);
      }
#pragma unroll
      for (int r = 0; r < 8; ++r) yacc[(mb + r) * kH + nt2 * 16 + n] = acc[r];
    }
    __syncthreads();
  }

  for (int i = tid; i < 16 * kH; i += 256) {
    int r = i >> 10, c = i & (kH - 1);
    int entry = toks[r];
    if (entry >= 0)
      contrib[(size_t)entry * kH + c] = gts[r] * (yacc[i] + b2[e * kH + c]);
  }
}

// ---------------- combine: out = x2 + slot contributions ----------------
__global__ __launch_bounds__(256) void combine_kernel(const float* __restrict__ x2,
                                                      const float* __restrict__ contrib,
                                                      float* __restrict__ out) {
  int t = blockIdx.x;
  for (int c = threadIdx.x; c < kH; c += 256) {
    size_t i = (size_t)t * kH + c;
    out[i] = x2[i] + contrib[(size_t)(t * 2) * kH + c]
                   + contrib[(size_t)(t * 2 + 1) * kH + c];
  }
}

// ---------------- host side ----------------
extern "C" void kernel_launch(void* const* d_in, const int* in_sizes, int n_in,
                              void* d_out, int out_size, void* d_ws, size_t ws_size,
                              hipStream_t stream) {
  const float* x     = (const float*)d_in[0];
  const float* ln1_g = (const float*)d_in[1];
  const float* ln1_b = (const float*)d_in[2];
  const float* qkv_w = (const float*)d_in[3];
  const float* qkv_b = (const float*)d_in[4];
  const float* out_w = (const float*)d_in[5];
  const float* out_b = (const float*)d_in[6];
  const float* ln2_g = (const float*)d_in[7];
  const float* ln2_b = (const float*)d_in[8];
  const float* Wg    = (const float*)d_in[9];
  const float* W1    = (const float*)d_in[10];
  const float* b1    = (const float*)d_in[11];
  const float* W2    = (const float*)d_in[12];
  const float* b2    = (const float*)d_in[13];
  float* out = (float*)d_out;

  char* ws = (char*)d_ws;
  size_t off = 0;
  auto carve = [&](size_t bytes) -> void* {
    void* p = ws + off;
    off = (off + bytes + 255) & ~(size_t)255;
    return p;
  };
  _Float16* hf16    = (_Float16*)carve((size_t)kT * kH * 2);
  _Float16* qkvf16  = (_Float16*)carve((size_t)kT * k3H * 2);
  _Float16* of16    = (_Float16*)carve((size_t)kT * kH * 2);
  float*    x2      = (float*)   carve((size_t)kT * kH * 4);
  _Float16* h2f16   = (_Float16*)carve((size_t)kT * kH * 2);
  int*      cnt     = (int*)     carve(kE * 4);
  int*      list    = (int*)     carve((size_t)kE * kT * 4);
  float*    gwbuf   = (float*)   carve((size_t)kE * kT * 4);
  float*    contrib = (float*)   carve((size_t)2 * kT * kH * 4);
  _Float16* qkvWh   = (_Float16*)carve((size_t)k3H * kH * 2);
  _Float16* outWh   = (_Float16*)carve((size_t)kH * kH * 2);
  _Float16* W1T     = (_Float16*)carve((size_t)kE * kH * kF * 2);
  _Float16* W2T     = (_Float16*)carve((size_t)kE * kF * kH * 2);

  // 0. one-time weight conversion / transposition to f16
  cvt_kernel<<<(k3H * kH) / 1024, 256, 0, stream>>>(qkv_w, qkvWh, k3H * kH);
  cvt_kernel<<<(kH * kH) / 1024, 256, 0, stream>>>(out_w, outWh, kH * kH);
  transpose_cvt_kernel<<<dim3(kF / 32, kH / 32, kE), 256, 0, stream>>>(
      W1, W1T, kH, kF);   // [H][F] -> [F][H]
  transpose_cvt_kernel<<<dim3(kH / 32, kF / 32, kE), 256, 0, stream>>>(
      W2, W2T, kF, kH);   // [F][H] -> [H][F]

  // 1. LN1
  ln_kernel<<<kT, 256, 0, stream>>>(x, ln1_g, ln1_b, hf16);
  // 2. QKV projection (f16 out)
  gemm_kernel<<<dim3(k3H / 32, kT / 64), 256, 0, stream>>>(
      hf16, qkvWh, qkv_b, nullptr, nullptr, qkvf16, kT, k3H, kH);
  // 3. attention
  size_t attn_smem = (size_t)16 * kS * 4 + (size_t)16 * kS * 2;
  attn_kernel<<<dim3(kS / 16, kNH, kB), 256, attn_smem, stream>>>(qkvf16, of16);
  // 4. output projection + residual(x) -> x2 (f32)
  gemm_kernel<<<dim3(kH / 32, kT / 64), 256, 0, stream>>>(
      of16, outWh, out_b, x, x2, nullptr, kT, kH, kH);
  // 5. LN2
  ln_kernel<<<kT, 256, 0, stream>>>(x2, ln2_g, ln2_b, h2f16);
  // 6. routing
  zero_cnt_kernel<<<1, 32, 0, stream>>>(cnt);
  router_kernel<<<kT, 256, 0, stream>>>(h2f16, Wg, cnt, list, gwbuf);
  // 7. fused MoE experts (top-2 only)
  size_t moe_smem = (size_t)16 * kH * 2 + (size_t)16 * kH * 4 + (size_t)16 * kFC * 2;
  moe_kernel<<<dim3(kT / 16, kE), 256, moe_smem, stream>>>(
      h2f16, W1T, b1, W2T, b2, cnt, list, gwbuf, contrib);
  // 8. combine + residual(x2)
  combine_kernel<<<kT, 256, 0, stream>>>(x2, contrib, out);
}
